// Model_24730421690434
// MI455X (gfx1250) — matrix operand order repaired and checked by reading, not compiled, not run on hardware
//
#include <hip/hip_runtime.h>
#include <hip/hip_bf16.h>

typedef __attribute__((ext_vector_type(16))) _Float16 v16h;
typedef __attribute__((ext_vector_type(8)))  _Float16 v8h;
typedef __attribute__((ext_vector_type(8)))  float    v8f;
typedef __attribute__((ext_vector_type(4)))  float    v4f;
typedef unsigned int u32x4 __attribute__((ext_vector_type(4)));
typedef int          i32x4 __attribute__((ext_vector_type(4)));
typedef int          i32x8 __attribute__((ext_vector_type(8)));

#define BATCH 512
#define SEQT  32
#define MSLOT 33
#define XK    1024   // P*P
#define ZD    128
#define KD    256
#define HD    512
#define G4    2048   // 4*H
#define KCAT  800    // H(512) + K+1(257) padded to 800

#ifndef __has_builtin
#define __has_builtin(x) 0
#endif
#if __has_builtin(__builtin_amdgcn_tensor_load_to_lds)
#define HAVE_TDM 1
#else
#define HAVE_TDM 0
#endif

// ---------------------------------------------------------------------------
// helpers
// ---------------------------------------------------------------------------
__device__ __forceinline__ v16h cat8h(v8h lo, v8h hi) {
  return __builtin_shufflevector(lo, hi, 0,1,2,3,4,5,6,7,8,9,10,11,12,13,14,15);
}
__device__ __forceinline__ float sigm(float x) { return 1.0f / (1.0f + __expf(-x)); }

#if HAVE_TDM
// Issue a 2D TDM tile load: tile_d0 (contiguous K) x tile_d1 (rows), f16 data,
// row padding 4 DWORDs after every 16 DWORDs -> LDS row stride 80B (bank-safe).
__device__ __forceinline__ void tdm_load_2d(void* lds_dst, const void* gsrc,
                                            unsigned tile_d0, unsigned tile_d1,
                                            unsigned long long stride0_elems) {
  unsigned long long ga = (unsigned long long)gsrc;
  unsigned lds_addr = (unsigned)(unsigned long long)lds_dst;  // low 32 = LDS offset
  u32x4 g0;
  g0[0] = 1u;                                         // count=1, user descriptor
  g0[1] = lds_addr;                                   // bits 63:32
  g0[2] = (unsigned)(ga & 0xffffffffu);               // global_addr[31:0]
  g0[3] = (unsigned)((ga >> 32) & 0x01ffffffu) | (2u << 30);  // ga[56:32] | type=2
  i32x8 g1;
  g1[0] = (int)((1u << 16) | (1u << 20) | (3u << 22) | (3u << 25));
          // data_size=1 (2B) | pad_enable | pad_interval=3 (16 dw) | pad_amount=3 (4 dw)
  g1[1] = (int)((tile_d0 & 0xffffu) << 16);           // tensor_dim0[15:0]  (== tile)
  g1[2] = (int)(((tile_d0 >> 16) & 0xffffu) | ((tile_d1 & 0xffffu) << 16)); // dim0 hi|dim1 lo
  g1[3] = (int)(((tile_d1 >> 16) & 0xffffu) | ((tile_d0 & 0xffffu) << 16)); // dim1 hi|tile_dim0
  g1[4] = (int)(tile_d1 & 0xffffu);                   // tile_dim1 | tile_dim2=0
  g1[5] = (int)(stride0_elems & 0xffffffffull);       // tensor_dim0_stride[31:0]
  g1[6] = (int)((stride0_elems >> 32) & 0xffffull);   // stride0[47:32] | stride1 lo=0
  g1[7] = 0;
  i32x4 z4 = {0, 0, 0, 0};
#if defined(__clang_major__) && __clang_major__ >= 23
  i32x8 z8 = {0, 0, 0, 0, 0, 0, 0, 0};
  __builtin_amdgcn_tensor_load_to_lds(g0, g1, z4, z4, z8, 0);
#else
  __builtin_amdgcn_tensor_load_to_lds(g0, g1, z4, z4, 0);
#endif
}
#endif  // HAVE_TDM

// ---------------------------------------------------------------------------
// prep: f16 weight copies, fused bias, zeroed activation / cell buffers
// ---------------------------------------------------------------------------
__global__ void k_prep(const float* __restrict__ W_enc, const float* __restrict__ W_ih,
                       const float* __restrict__ W_hh, const float* __restrict__ b_ih,
                       const float* __restrict__ b_hh, const float* __restrict__ W_key,
                       _Float16* __restrict__ Wcat, _Float16* __restrict__ WencT,
                       _Float16* __restrict__ Wkey, float* __restrict__ bias,
                       _Float16* __restrict__ Acat, float* __restrict__ cbuf) {
  long i = (long)blockIdx.x * 256 + threadIdx.x;
  if (i < (long)G4 * KCAT) {                       // Wcat[j][col]
    int j = (int)(i / KCAT), col = (int)(i % KCAT);
    float v = 0.0f;
    if (col < HD)            v = W_hh[(long)j * HD + col];
    else if (col < HD + 257) v = W_ih[(long)j * 257 + (col - HD)];
    Wcat[i] = (_Float16)v;
    return;
  }
  i -= (long)G4 * KCAT;
  if (i < ZD * XK) {                               // WencT[n][k] = W_enc[k][n]
    int n = (int)(i >> 10), k = (int)(i & 1023);
    WencT[i] = (_Float16)W_enc[(long)k * ZD + n];
    return;
  }
  i -= ZD * XK;
  if (i < KD * HD) { Wkey[i] = (_Float16)W_key[i]; return; }
  i -= KD * HD;
  if (i < G4) { bias[i] = b_ih[i] + b_hh[i]; return; }
  i -= G4;
  if (i < (long)BATCH * KCAT) { Acat[i] = (_Float16)0.0f; return; }
  i -= (long)BATCH * KCAT;
  if (i < (long)BATCH * HD) { cbuf[i] = 0.0f; return; }
}

// ---------------------------------------------------------------------------
// encoder: z = Xflat[16384,1024] @ W_enc[1024,128]  (WMMA f16, f32 accum)
// wave computes 16x64 (4 tiles) reusing the converted A fragment 4x.
// block = 8 waves = 4 row-bands x 2 col-halves -> 64 rows, 128 cols. grid=256
// ---------------------------------------------------------------------------
__global__ void k_encoder(const float* __restrict__ X, const _Float16* __restrict__ WencT,
                          float* __restrict__ z_pad) {
  const int lane = threadIdx.x & 31, wv = threadIdx.x >> 5;
  const int band = wv >> 1, half = wv & 1;
  const int r0 = blockIdx.x * 64 + band * 16;
  const int c0 = half * 64;
  const int rA  = lane & 15;
  const int kbA = (lane < 16) ? 0 : 8;
  const int kbB = (lane < 16) ? 0 : 8;
  const float* ap = X + (long)(r0 + rA) * XK + kbA;
  v8f acc[4] = {};
  for (int kc = 0; kc < XK; kc += 32) {
    __builtin_prefetch(ap + kc + 256, 0, 1);
    v4f x0 = *(const v4f*)(ap + kc);
    v4f x1 = *(const v4f*)(ap + kc + 4);
    v4f x2 = *(const v4f*)(ap + kc + 16);
    v4f x3 = *(const v4f*)(ap + kc + 20);
    v16h a;
#pragma unroll
    for (int q = 0; q < 4; ++q) {
      a[q]      = (_Float16)x0[q];
      a[4 + q]  = (_Float16)x1[q];
      a[8 + q]  = (_Float16)x2[q];
      a[12 + q] = (_Float16)x3[q];
    }
#pragma unroll
    for (int tt = 0; tt < 4; ++tt) {
      const _Float16* bp = WencT + (long)(c0 + tt * 16 + rA) * XK + kc + kbB;
      v16h b = cat8h(*(const v8h*)bp, *(const v8h*)(bp + 16));
      acc[tt] = __builtin_amdgcn_wmma_f32_16x16x32_f16(false, a, false, b, (short)0,
                                                       acc[tt], false, false);
    }
  }
  const int ro = (lane < 16) ? 0 : 8;
#pragma unroll
  for (int tt = 0; tt < 4; ++tt) {
    int col = c0 + tt * 16 + rA;
#pragma unroll
    for (int q = 0; q < 8; ++q) {
      int r = r0 + ro + q;
      int b = r >> 5, t = r & 31;
      z_pad[((long)b * MSLOT + t) * ZD + col] = acc[tt][q];
    }
  }
}

// ---------------------------------------------------------------------------
// context norm over T per (b, channel); zero the final (t=32) slot
// ---------------------------------------------------------------------------
__global__ void k_norm(float* __restrict__ z_pad, const float* __restrict__ gamma,
                       const float* __restrict__ beta) {
  int i = blockIdx.x * 256 + threadIdx.x;   // b*128 + zc, 65536 total
  int b = i >> 7, zc = i & 127;
  float* base = z_pad + (long)b * MSLOT * ZD + zc;
  float s = 0.0f, s2 = 0.0f;
#pragma unroll 4
  for (int t = 0; t < SEQT; ++t) { float v = base[t * ZD]; s += v; s2 += v * v; }
  float mu  = s * (1.0f / SEQT);
  float var = s2 * (1.0f / SEQT) - mu * mu;
  float inv = rsqrtf(var + 1e-8f);
  float ga = gamma[zc], be = beta[zc];
#pragma unroll 4
  for (int t = 0; t < SEQT; ++t) base[t * ZD] = (base[t * ZD] - mu) * inv * ga + be;
  base[SEQT * ZD] = 0.0f;
}

// ---------------------------------------------------------------------------
// gates = Acat[512,800] @ Wcat[2048,800]^T + bias  (WMMA)
// block = 128 threads (4 waves) covering 16 rows x 256 cols; per k-chunk the
// 256x32 Wcat tile is staged into LDS by the Tensor Data Mover (double-
// buffered, TDM row padding -> 80B LDS rows, bank-conflict-free ds_load_b128).
// wave computes 16x64 = 4 WMMAs per chunk, A fragment reused 4x. Fully
// unrolled so the 25 chunks software-pipeline across barriers.
// grid = 32 rowTiles x 8 colGroups = 256 blocks.
// ---------------------------------------------------------------------------
#define LROW 40   // padded LDS row stride in halfs (80 bytes)
__global__ void k_gates(const _Float16* __restrict__ Acat, const _Float16* __restrict__ Wcat,
                        const float* __restrict__ bias, float* __restrict__ gates) {
  __shared__ _Float16 smem[2][256 * LROW];          // 2 x 20 KB
  const int lane = threadIdx.x & 31, wv = threadIdx.x >> 5;   // wv = 0..3
  const int rowTile = blockIdx.x >> 3;              // 0..31
  const int cg = blockIdx.x & 7;                    // 0..7 -> 256-col group
  const int r0 = rowTile * 16;
  const int rA  = lane & 15;
  const int kbA = (lane < 16) ? 0 : 8;
  const int kbB = (lane < 16) ? 0 : 8;
  const _Float16* ap = Acat + (long)(r0 + rA) * KCAT + kbA;
  const _Float16* wbase = Wcat + (long)cg * 256 * KCAT;
  v8f acc[4] = {};

#if HAVE_TDM
  if (wv == 0) tdm_load_2d(&smem[0][0], wbase, 32, 256, KCAT);
#endif
#pragma unroll
  for (int i = 0; i < 25; ++i) {
    const int kc = i * 32;
    _Float16* cur = &smem[i & 1][0];
#if HAVE_TDM
    if (wv == 0) __builtin_amdgcn_s_wait_tensorcnt(0);
    __syncthreads();                                 // cur buffer ready for all
    if (wv == 0 && i + 1 < 25)
      tdm_load_2d(&smem[(i + 1) & 1][0], wbase + kc + 32, 32, 256, KCAT);
#else
    __syncthreads();
    for (int rr = threadIdx.x; rr < 256; rr += 128) {  // cooperative fallback
      const _Float16* s = wbase + (long)rr * KCAT + kc;
      _Float16* d = cur + rr * LROW;
#pragma unroll
      for (int q = 0; q < 4; ++q) *(v8h*)(d + q * 8) = *(const v8h*)(s + q * 8);
    }
    __syncthreads();
#endif
    v8h a0 = *(const v8h*)(ap + kc);
    v8h a1 = *(const v8h*)(ap + kc + 16);
    v16h a = cat8h(a0, a1);
#pragma unroll
    for (int tt = 0; tt < 4; ++tt) {
      const _Float16* bp = cur + (wv * 64 + tt * 16 + rA) * LROW + kbB;
      v16h b = cat8h(*(const v8h*)bp, *(const v8h*)(bp + 16));
      acc[tt] = __builtin_amdgcn_wmma_f32_16x16x32_f16(false, a, false, b, (short)0,
                                                       acc[tt], false, false);
    }
    __syncthreads();                                 // done reading cur
  }
  const int ro = (lane < 16) ? 0 : 8;
#pragma unroll
  for (int tt = 0; tt < 4; ++tt) {
    int col = cg * 256 + wv * 64 + tt * 16 + rA;
    float bb = bias[col];
#pragma unroll
    for (int q = 0; q < 8; ++q)
      gates[(long)(r0 + ro + q) * G4 + col] = acc[tt][q] + bb;
  }
}

// ---------------------------------------------------------------------------
// LSTM cell pointwise: c,h update; h also written as f16 into Acat[:,0:512]
// ---------------------------------------------------------------------------
__global__ void k_cell(const float* __restrict__ gates, float* __restrict__ cbuf,
                       float* __restrict__ hbuf, _Float16* __restrict__ Acat) {
  int i = blockIdx.x * 256 + threadIdx.x;   // b*512 + j
  int b = i >> 9, j = i & 511;
  const float* gr = gates + (long)b * G4;
  float gi = gr[j], gf = gr[HD + j], gg = gr[2 * HD + j], go = gr[3 * HD + j];
  float c = sigm(gf) * cbuf[i] + sigm(gi) * tanhf(gg);
  float h = sigm(go) * tanhf(c);
  cbuf[i] = c;
  hbuf[i] = h;
  Acat[(long)b * KCAT + j] = (_Float16)h;
}

// ---------------------------------------------------------------------------
// key_w = relu(h @ W_key^T + b_key) -> written straight into M_k slot t (WMMA)
// ---------------------------------------------------------------------------
__global__ void k_keyw(const _Float16* __restrict__ Acat, const _Float16* __restrict__ Wkey,
                       const float* __restrict__ b_key, float* __restrict__ Mk, int t) {
  const int lane = threadIdx.x & 31, w = threadIdx.x >> 5;
  const int rowTile = blockIdx.x >> 1;
  const int colTile = ((blockIdx.x & 1) << 3) + w;
  const int r0 = rowTile * 16, c0 = colTile * 16;
  const int rA  = lane & 15;
  const int kbA = (lane < 16) ? 0 : 8;
  const int kbB = (lane < 16) ? 0 : 8;
  const _Float16* ap = Acat + (long)(r0 + rA) * KCAT + kbA;   // h region = cols 0..511
  const _Float16* bp = Wkey + (long)(c0 + rA) * HD + kbB;
  v8f acc = {};
#pragma unroll
  for (int kc = 0; kc < HD; kc += 32) {
    v8h a0 = *(const v8h*)(ap + kc);
    v8h a1 = *(const v8h*)(ap + kc + 16);
    v8h b0 = *(const v8h*)(bp + kc);
    v8h b1 = *(const v8h*)(bp + kc + 16);
    acc = __builtin_amdgcn_wmma_f32_16x16x32_f16(false, cat8h(a0, a1), false,
                                                 cat8h(b0, b1), (short)0, acc,
                                                 false, false);
  }
  const int col = c0 + rA;
  const int ro  = (lane < 16) ? 0 : 8;
#pragma unroll
  for (int q = 0; q < 8; ++q) {
    int b = r0 + ro + q;
    Mk[((long)b * MSLOT + t) * KD + col] = fmaxf(acc[q] + b_key[col], 0.0f);
  }
}

// ---------------------------------------------------------------------------
// attention read + gate g + key_r update (one block per batch row)
// ---------------------------------------------------------------------------
__global__ void k_attn(const float* __restrict__ z_pad, const float* __restrict__ Mk,
                       const float* __restrict__ hbuf, const float* __restrict__ W_g,
                       const float* __restrict__ b_g, const float* __restrict__ conf_gain,
                       const float* __restrict__ conf_bias, _Float16* __restrict__ Acat,
                       int t) {
  const int b = blockIdx.x, tid = threadIdx.x;
  __shared__ float s_red[256];
  __shared__ float s_wk[MSLOT];
  __shared__ float s_ck[MSLOT];
  __shared__ float s_g, s_inv;

  // ---- g = sigmoid(h . W_g + b_g)
  float p = hbuf[(long)b * HD + tid] * W_g[tid] +
            hbuf[(long)b * HD + 256 + tid] * W_g[256 + tid];
  s_red[tid] = p;
  __syncthreads();
  for (int s = 128; s > 0; s >>= 1) {
    if (tid < s) s_red[tid] += s_red[tid + s];
    __syncthreads();
  }
  if (tid == 0) s_g = sigm(s_red[0] + b_g[0]);

  // ---- scores[m] = z_t . z_pad[m], m < t  (wave per slot, lane-parallel dot)
  const int lane = tid & 31, w = tid >> 5;
  const float* zt = z_pad + ((long)b * MSLOT + t) * ZD;
  for (int m = w; m < t; m += 8) {
    const float* zm = z_pad + ((long)b * MSLOT + m) * ZD;
    float4 a = *(const float4*)(zt + lane * 4);
    float4 c = *(const float4*)(zm + lane * 4);
    float d = a.x * c.x + a.y * c.y + a.z * c.z + a.w * c.w;
#pragma unroll
    for (int s = 16; s > 0; s >>= 1) d += __shfl_xor(d, s, 32);
    if (lane == 0) s_wk[m] = d;
  }
  __syncthreads();

  // ---- softmax over valid slots + confidence c_k
  if (tid == 0 && t > 0) {
    float mx = -1e30f;
    for (int m = 0; m < t; ++m) mx = fmaxf(mx, s_wk[m]);
    float sum = 0.0f;
    for (int m = 0; m < t; ++m) {
      float sc = s_wk[m];
      s_ck[m] = sigm(sc * conf_gain[0] + conf_bias[0]);
      float e = __expf(sc - mx);
      s_wk[m] = e;
      sum += e;
    }
    s_inv = 1.0f / sum;
  }
  __syncthreads();

  // ---- key_r = g * sum_m w[m] * [M_k | c_k]  -> Acat cols 512..768 (f16)
  _Float16* kr = Acat + (long)b * KCAT + HD;
  if (t == 0) {
    kr[tid] = (_Float16)0.0f;
    if (tid == 0) kr[256] = (_Float16)0.0f;
  } else {
    float acc = 0.0f;
    for (int m = 0; m < t; ++m)
      acc += s_wk[m] * Mk[((long)b * MSLOT + m) * KD + tid];
    kr[tid] = (_Float16)(s_g * s_inv * acc);
    if (tid == 0) {
      float a2 = 0.0f;
      for (int m = 0; m < t; ++m) a2 += s_wk[m] * s_ck[m];
      kr[256] = (_Float16)(s_g * s_inv * a2);
    }
  }
}

// ---------------------------------------------------------------------------
// head: y = h @ W_y^T + b_y ; argmax.  out[0:2048]=y_lin, out[2048:2560]=argmax
// ---------------------------------------------------------------------------
__global__ void k_final(const float* __restrict__ hbuf, const float* __restrict__ W_y,
                        const float* __restrict__ b_y, float* __restrict__ out) {
  const int b = blockIdx.x, tid = threadIdx.x;  // 128 threads
  const int y = tid >> 5, lane = tid & 31;
  const float* h  = hbuf + (long)b * HD;
  const float* wy = W_y + (long)y * HD;
  float p = 0.0f;
  for (int q = lane; q < HD; q += 32) p += h[q] * wy[q];
#pragma unroll
  for (int s = 16; s > 0; s >>= 1) p += __shfl_xor(p, s, 32);
  __shared__ float sy[4];
  if (lane == 0) sy[y] = p + b_y[y];
  __syncthreads();
  if (tid == 0) {
    float best = sy[0];
    int bi = 0;
    for (int q = 1; q < 4; ++q)
      if (sy[q] > best) { best = sy[q]; bi = q; }
    for (int q = 0; q < 4; ++q) out[b * 4 + q] = sy[q];
    out[2048 + b] = (float)bi;
  }
}

// ---------------------------------------------------------------------------
// host-side orchestration
// ---------------------------------------------------------------------------
extern "C" void kernel_launch(void* const* d_in, const int* in_sizes, int n_in,
                              void* d_out, int out_size, void* d_ws, size_t ws_size,
                              hipStream_t stream) {
  const float* x_seq     = (const float*)d_in[0];
  const float* W_enc     = (const float*)d_in[1];
  const float* gamma     = (const float*)d_in[2];
  const float* beta      = (const float*)d_in[3];
  const float* W_ih      = (const float*)d_in[4];
  const float* W_hh      = (const float*)d_in[5];
  const float* b_ih      = (const float*)d_in[6];
  const float* b_hh      = (const float*)d_in[7];
  const float* W_key     = (const float*)d_in[8];
  const float* b_key     = (const float*)d_in[9];
  const float* W_g       = (const float*)d_in[10];
  const float* b_g       = (const float*)d_in[11];
  const float* conf_gain = (const float*)d_in[12];
  const float* conf_bias = (const float*)d_in[13];
  const float* W_y       = (const float*)d_in[14];
  const float* b_y       = (const float*)d_in[15];
  float* out = (float*)d_out;

  char* ws = (char*)d_ws;
  size_t off = 0;
  auto carve = [&](size_t bytes) -> void* {
    void* p = ws + off;
    off = (off + bytes + 255) & ~(size_t)255;
    return p;
  };
  float*    z_pad = (float*)   carve((size_t)BATCH * MSLOT * ZD * 4);   // 8.65 MB
  _Float16* Wcat  = (_Float16*)carve((size_t)G4 * KCAT * 2);            // 3.28 MB
  _Float16* WencT = (_Float16*)carve((size_t)ZD * XK * 2);              // 0.26 MB
  _Float16* Wkey  = (_Float16*)carve((size_t)KD * HD * 2);              // 0.26 MB
  float*    bias  = (float*)   carve((size_t)G4 * 4);
  _Float16* Acat  = (_Float16*)carve((size_t)BATCH * KCAT * 2);         // 0.82 MB
  float*    gates = (float*)   carve((size_t)BATCH * G4 * 4);           // 4.19 MB
  float*    cbuf  = (float*)   carve((size_t)BATCH * HD * 4);           // 1.05 MB
  float*    hbuf  = (float*)   carve((size_t)BATCH * HD * 4);           // 1.05 MB
  float*    Mk    = (float*)   carve((size_t)BATCH * MSLOT * KD * 4);   // 17.3 MB

  // prep (weights -> f16, fused bias, zero state)
  {
    long total = (long)G4 * KCAT + ZD * XK + KD * HD + G4 +
                 (long)BATCH * KCAT + (long)BATCH * HD;
    int grid = (int)((total + 255) / 256);
    k_prep<<<grid, 256, 0, stream>>>(W_enc, W_ih, W_hh, b_ih, b_hh, W_key,
                                     Wcat, WencT, Wkey, bias, Acat, cbuf);
  }
  // encoder GEMM + context norm
  k_encoder<<<256, 256, 0, stream>>>(x_seq, WencT, z_pad);
  k_norm<<<256, 256, 0, stream>>>(z_pad, gamma, beta);

  // 33 recurrent steps
  for (int t = 0; t < MSLOT; ++t) {
    k_gates<<<256, 128, 0, stream>>>(Acat, Wcat, bias, gates);
    k_cell<<<1024, 256, 0, stream>>>(gates, cbuf, hbuf, Acat);
    if (t < MSLOT - 1) {
      k_keyw<<<64, 256, 0, stream>>>(Acat, Wkey, b_key, Mk, t);
      k_attn<<<512, 256, 0, stream>>>(z_pad, Mk, hbuf, W_g, b_g, conf_gain,
                                      conf_bias, Acat, t);
    }
  }
  // head + argmax
  k_final<<<512, 128, 0, stream>>>(hbuf, W_y, b_y, out);
}
